// RNNModel_40063454937639
// MI455X (gfx1250) — matrix operand order repaired and checked by reading, not compiled, not run on hardware
//
#include <hip/hip_runtime.h>
#include <cstdint>
#include <cstddef>

// ---------------------------------------------------------------------------
// 2-layer bidirectional LSTM for MI455X (gfx1250, wave32, WMMA f16->f32).
//   B=32, T=2048, D_INPUT=256, D_HIDDEN=256, gates order i,f,g,o.
// Phase 1: xw = x @ w_ih^T + b  -- WMMA GEMM; x[t] staged into LDS by the
//          Tensor Data Mover (tensor_load_to_lds, D# per ISA ch.8), B tiles
//          chunk-preloaded for memory/WMMA overlap. Output stored in WMMA
//          C-tile swizzled f16 layout.
// Phase 2: sequential scan, 1 workgroup (32 waves) per (layer, direction);
//          w_hh register-resident as WMMA B-tiles, h in LDS (padded rows),
//          c in registers, symmetric gate exchange via LDS (all 32 waves do
//          elementwise work), V_TANH_F32 for all activations, 2 barriers/step.
// ---------------------------------------------------------------------------

typedef __attribute__((ext_vector_type(16))) _Float16 v16h;
typedef __attribute__((ext_vector_type(8)))  _Float16 v8h;
typedef __attribute__((ext_vector_type(8)))  float    v8f;
typedef __attribute__((ext_vector_type(4)))  uint32_t su4;
typedef __attribute__((ext_vector_type(8)))  uint32_t su8;

#define T_SEQ   2048
#define BATCH   32
#define HID     256
#define G4      1024           // 4*HID
#define HPAD    264            // padded LDS row stride (halfs) -> no bank conflicts
#define XROWB   1040           // staged x row stride in bytes (1024 data + 16 pad)

__device__ __forceinline__ v8f wmma_f16(v16h a, v16h b, v8f c) {
  return __builtin_amdgcn_wmma_f32_16x16x32_f16(false, a, false, b, (short)0, c,
                                                false, false);
}

// load 16 halfs (two 16B chunks) into one v16h A/B operand
__device__ __forceinline__ v16h ld2x16h(const _Float16* p0, const _Float16* p1) {
  v16h r;
  uint4* q = reinterpret_cast<uint4*>(&r);
  q[0] = *reinterpret_cast<const uint4*>(p0);
  q[1] = *reinterpret_cast<const uint4*>(p1);
  return r;
}

// gfx1250 hardware tanh (TRANS op). Trailing v_nop satisfies the ISA TRANS
// hazard rule (1 independent op / V_NOP before the result may be consumed).
__device__ __forceinline__ float tanh_hw(float x) {
  float r;
  asm volatile("v_tanh_f32 %0, %1\n\tv_nop" : "=v"(r) : "v"(x));
  return r;
}
__device__ __forceinline__ float sig_hw(float x) {
  return __builtin_fmaf(tanh_hw(0.5f * x), 0.5f, 0.5f);
}

// -------------------------------- weight convert ---------------------------
__global__ void cvt_f32_to_f16(const float* __restrict__ s, _Float16* __restrict__ d,
                               int n) {
  int i = blockIdx.x * blockDim.x + threadIdx.x;
  if (i < n) d[i] = (_Float16)s[i];
}

// -------------------------------- phase 1: xw GEMM -------------------------
// grid: (T, 2 dirs), block: 256 (8 waves). wave -> (mt = wave&1, 16 n-tiles).
// x[t] (32 x DIN) is DMA'd into LDS once per block by the TDM; rows are padded
// by 16B every 1024B (pad_interval=7, pad_amount=3) -> stride 1040B, so the
// per-row ds_load_b128 A-tile reads step 4 banks per lane (conflict-free).
template <typename XT, int DIN>
__global__ __launch_bounds__(256) void gemm_xw_kernel(
    const XT* __restrict__ x,            // [T][32][DIN]
    const _Float16* __restrict__ wih,    // [2][1024][DIN] f16
    const float* __restrict__ bias_f,    // [1024]
    const float* __restrict__ bias_b,    // [1024]
    _Float16* __restrict__ xw)           // [2][T*2*64*256] halfs
{
  extern __shared__ char smem[];         // staged x[t]: 32 rows * 1040B
  const int t    = blockIdx.x;
  const int dir  = blockIdx.y;
  const int wave = threadIdx.x >> 5;
  const int lane = threadIdx.x & 31;
  const int l16  = lane & 15;
  const int hf   = lane >> 4;            // lane half
  const int mt   = wave & 1;
  constexpr int KT = DIN / 32;

  const _Float16* w    = wih + (size_t)dir * G4 * DIN;
  const float*    bias = dir ? bias_b : bias_f;
  _Float16*       xwd  = xw + (size_t)dir * ((size_t)T_SEQ * 2 * 64 * 256);

  // ---- TDM: DMA x[t] into LDS (wave 0 issues; D# built per ISA ch.8) ----
  if (threadIdx.x < 32) {
    const uint64_t ga   = (uint64_t)(const void*)(x + (size_t)t * BATCH * DIN);
    const uint32_t lds0 = __builtin_amdgcn_groupstaticsize(); // dyn-LDS base
    constexpr uint32_t dsz = (sizeof(XT) == 4) ? 2u : 1u;     // 4B / 2B code
    su4 g0;
    g0[0] = 1u;                                           // count=1, flags=0
    g0[1] = lds0;                                         // lds_addr
    g0[2] = (uint32_t)ga;                                 // global_addr lo
    g0[3] = (uint32_t)((ga >> 32) & 0x01FFFFFFu) | (2u << 30); // hi | type=2
    su8 g1;
    g1[0] = (dsz << 16) | (1u << 20) | (7u << 22) | (3u << 25);
    //        data_size    pad_enable   interval=256dw  amount=4dw(16B)
    g1[1] = ((uint32_t)DIN & 0xFFFFu) << 16;              // tensor_dim0 lo16
    g1[2] = ((uint32_t)DIN >> 16) | ((uint32_t)BATCH << 16); // dim0 hi | dim1 lo
    g1[3] = ((uint32_t)DIN) << 16;                        // dim1 hi=0 | tile_dim0
    g1[4] = (uint32_t)BATCH;                              // tile_dim1 | tile_dim2=0
    g1[5] = (uint32_t)DIN;                                // dim0_stride lo32
    g1[6] = 0u;                                           // stride0 hi | stride1 lo
    g1[7] = 0u;
    asm volatile("tensor_load_to_lds %0, %1, null, null"
                 :: "s"(g0), "s"(g1) : "memory");
    __builtin_amdgcn_s_wait_tensorcnt(0);
  }
  __syncthreads();

  // ---- load A tiles (this wave's 16 rows) from LDS, convert to f16 ----
  v16h A[KT];
  const int m = mt * 16 + l16;
  const char* xrow = smem + (size_t)m * XROWB;
  for (int kt = 0; kt < KT; ++kt) {
    const int kb = kt * 32 + hf * 8;
    if constexpr (sizeof(XT) == 4) {
      const float* src = reinterpret_cast<const float*>(xrow) + kb;
      float tmp[16];
      *reinterpret_cast<float4*>(tmp + 0)  = *reinterpret_cast<const float4*>(src + 0);
      *reinterpret_cast<float4*>(tmp + 4)  = *reinterpret_cast<const float4*>(src + 4);
      *reinterpret_cast<float4*>(tmp + 8)  = *reinterpret_cast<const float4*>(src + 16);
      *reinterpret_cast<float4*>(tmp + 12) = *reinterpret_cast<const float4*>(src + 20);
      v16h a;
      for (int i = 0; i < 16; ++i) a[i] = (_Float16)tmp[i];
      A[kt] = a;
    } else {
      const _Float16* src = reinterpret_cast<const _Float16*>(xrow) + kb;
      A[kt] = ld2x16h(src, src + 16);
    }
  }

  // ---- 16 n-tiles per wave; B preloaded in chunks of 8 tiles for MLP ----
  for (int nti = 0; nti < 16; ++nti) {
    const int nt = (wave >> 1) + nti * 4;
    const int n  = nt * 16 + l16;
    const float bv = bias[n];
    v8f acc;
    for (int r = 0; r < 8; ++r) acc[r] = bv;
    for (int kc = 0; kc < KT; kc += 8) {
      v16h bt[8];
#pragma unroll
      for (int u = 0; u < 8; ++u) {
        const _Float16* bp = w + (size_t)n * DIN + (kc + u) * 32 + hf * 8;
        bt[u] = ld2x16h(bp, bp + 16);
      }
#pragma unroll
      for (int u = 0; u < 8; ++u) acc = wmma_f16(A[kc + u], bt[u], acc);
    }
    v8h o;
    for (int r = 0; r < 8; ++r) o[r] = (_Float16)acc[r];
    _Float16* dst = xwd + (((size_t)t * 2 + mt) * 64 + nt) * 256 + lane * 8;
    *reinterpret_cast<uint4*>(dst) = *reinterpret_cast<uint4*>(&o);
  }
}

// -------------------------------- phase 2: recurrent scan ------------------
// grid: 2 (dir), block: 1024 (32 waves).
// wave w<16  : gates i,f for hidden tile w    (cols w*16 / 256+w*16)
// wave w>=16 : gates g,o for hidden tile w-16 (cols 512+.. / 768+..)
// Exchange: i/f-wave publishes its mt=1 tiles, g/o-wave its mt=0 tiles;
// i/f-wave updates rows 0..15, g/o-wave rows 16..31 (all waves busy).
template <int LAYER>
__global__ __launch_bounds__(1024) void lstm_scan_kernel(
    const _Float16* __restrict__ xw,     // swizzled [2][T*2*64*256]
    const _Float16* __restrict__ whh16,  // [4][1024][256] f16 (2l+d)
    _Float16* __restrict__ y0,           // layer0 out: [T][32][512] f16
    float* __restrict__ out,             // layer1 out: [32][T][512] f32
    float* __restrict__ hn,              // [4][32][256]
    float* __restrict__ cn)              // [4][32][256]
{
  extern __shared__ char smem[];
  _Float16* hbuf = reinterpret_cast<_Float16*>(smem);                // [32][HPAD]
  float*    gob  = reinterpret_cast<float*>(smem + BATCH * HPAD * 2); // 64KB

  const int dir  = blockIdx.x;
  const int wave = threadIdx.x >> 5;
  const int lane = threadIdx.x & 31;
  const int l16  = lane & 15;
  const int hf   = lane >> 4;
  const bool isIF = (wave < 16);
  const int p    = isIF ? wave : wave - 16;
  int nc[2];
  nc[0] = (isIF ? 0 : 512) + p * 16;     // gate i or g
  nc[1] = nc[0] + 256;                   // gate f or o

  const _Float16* w   = whh16 + (size_t)(LAYER * 2 + dir) * G4 * HID;
  const _Float16* xwd = xw + (size_t)dir * ((size_t)T_SEQ * 2 * 64 * 256);

  // ---- register-resident w_hh B-tiles: 2 gate-cols x 8 k-tiles ----
  v16h Bt[2][8];
  for (int g = 0; g < 2; ++g)
    for (int kt = 0; kt < 8; ++kt) {
      const int n  = nc[g] + l16;
      const int k0 = kt * 32 + hf * 8;
      const _Float16* bp = w + (size_t)n * HID + k0;
      Bt[g][kt] = ld2x16h(bp, bp + 16);
    }

  float cst[8], hlast[8];
  for (int i = 0; i < 8; ++i) { cst[i] = 0.0f; hlast[i] = 0.0f; }

  for (int i = threadIdx.x; i < BATCH * HPAD; i += 1024) hbuf[i] = (_Float16)0;
  __syncthreads();

  const _Float16* hrow0 = hbuf + (size_t)l16 * HPAD;
  const _Float16* hrow1 = hbuf + (size_t)(16 + l16) * HPAD;

  // exchange slots (per pair p, 1024 floats):
  //   +0   : g (mt0)   written by g/o-wave      +256 : o (mt0)
  //   +512 : i (mt1)   written by i/f-wave      +768 : f (mt1)
  float* pubbase  = gob + (size_t)p * 1024 + (isIF ? 512 : 0);
  const float* rd = gob + (size_t)p * 1024 + (isIF ? 0 : 512);
  const int mloc  = isIF ? 0 : 1;          // which m-half this wave updates
  const int mrow0 = mloc * 16 + hf * 8;    // first of its 8 rows
  const int j     = p * 16 + l16;

  for (int s = 0; s < T_SEQ; ++s) {
    const int t = dir ? (T_SEQ - 1 - s) : s;

    // accumulators init from precomputed xw tiles (already include bias)
    v8f acc[2][2];
    for (int g = 0; g < 2; ++g)
      for (int mt = 0; mt < 2; ++mt) {
        const _Float16* xp =
            xwd + (((size_t)t * 2 + mt) * 64 + (nc[g] >> 4)) * 256 + lane * 8;
        v8h xv = *reinterpret_cast<const v8h*>(xp);
        for (int r = 0; r < 8; ++r) acc[g][mt][r] = (float)xv[r];
      }

    // prefetch next timestep's xw tiles into cache
    if (s + 1 < T_SEQ) {
      const int tn = dir ? (t - 1) : (t + 1);
      for (int g = 0; g < 2; ++g)
        for (int mt = 0; mt < 2; ++mt)
          __builtin_prefetch(
              xwd + (((size_t)tn * 2 + mt) * 64 + (nc[g] >> 4)) * 256 + lane * 8,
              0, 1);
    }

    // g += h @ w_hh^T : 8 k-steps, A loads software-pipelined one step ahead
    {
      const int kb = hf * 8;
      v16h A0 = ld2x16h(hrow0 + kb, hrow0 + kb + 16);
      v16h A1 = ld2x16h(hrow1 + kb, hrow1 + kb + 16);
#pragma unroll
      for (int kt = 0; kt < 8; ++kt) {
        v16h nA0, nA1;
        if (kt < 7) {
          const int kbn = (kt + 1) * 32 + hf * 8;
          nA0 = ld2x16h(hrow0 + kbn, hrow0 + kbn + 16);
          nA1 = ld2x16h(hrow1 + kbn, hrow1 + kbn + 16);
        }
        acc[0][0] = wmma_f16(A0, Bt[0][kt], acc[0][0]);
        acc[0][1] = wmma_f16(A1, Bt[0][kt], acc[0][1]);
        acc[1][0] = wmma_f16(A0, Bt[1][kt], acc[1][0]);
        acc[1][1] = wmma_f16(A1, Bt[1][kt], acc[1][1]);
        A0 = nA0; A1 = nA1;
      }
    }

    // publish the 2 tiles the partner wave needs
    if (isIF) {
      const float4* a0 = reinterpret_cast<const float4*>(&acc[0][1]);
      const float4* a1 = reinterpret_cast<const float4*>(&acc[1][1]);
      reinterpret_cast<float4*>(pubbase + lane * 8)[0] = a0[0];
      reinterpret_cast<float4*>(pubbase + lane * 8)[1] = a0[1];
      reinterpret_cast<float4*>(pubbase + 256 + lane * 8)[0] = a1[0];
      reinterpret_cast<float4*>(pubbase + 256 + lane * 8)[1] = a1[1];
    } else {
      const float4* a0 = reinterpret_cast<const float4*>(&acc[0][0]);
      const float4* a1 = reinterpret_cast<const float4*>(&acc[1][0]);
      reinterpret_cast<float4*>(pubbase + lane * 8)[0] = a0[0];
      reinterpret_cast<float4*>(pubbase + lane * 8)[1] = a0[1];
      reinterpret_cast<float4*>(pubbase + 256 + lane * 8)[0] = a1[0];
      reinterpret_cast<float4*>(pubbase + 256 + lane * 8)[1] = a1[1];
    }
    __syncthreads();

    // every wave updates its 8 rows; c stays in registers the whole scan
    for (int r = 0; r < 8; ++r) {
      float iv, fv, gv, ov;
      if (isIF) {               // own i,f (mt0); partner g,o from LDS
        iv = acc[0][0][r];
        fv = acc[1][0][r];
        gv = rd[lane * 8 + r];
        ov = rd[256 + lane * 8 + r];
      } else {                  // partner i,f from LDS; own g,o (mt1)
        iv = rd[lane * 8 + r];
        fv = rd[256 + lane * 8 + r];
        gv = acc[0][1][r];
        ov = acc[1][1][r];
      }
      float c   = sig_hw(fv) * cst[r] + sig_hw(iv) * tanh_hw(gv);
      cst[r]    = c;
      const float hv = sig_hw(ov) * tanh_hw(c);
      hlast[r]  = hv;
      const int mrow = mrow0 + r;
      hbuf[(size_t)mrow * HPAD + j] = (_Float16)hv;
      if constexpr (LAYER == 0) {
        y0[((size_t)t * BATCH + mrow) * 512 + dir * 256 + j] = (_Float16)hv;
      } else {
        out[((size_t)mrow * T_SEQ + t) * 512 + dir * 256 + j] = hv;
      }
    }
    __syncthreads();
  }

  for (int r = 0; r < 8; ++r) {
    const int mrow = mrow0 + r;
    const size_t o = ((size_t)(2 * LAYER + dir) * BATCH + mrow) * HID + j;
    hn[o] = hlast[r];
    cn[o] = cst[r];
  }
}

// -------------------------------- launch -----------------------------------
extern "C" void kernel_launch(void* const* d_in, const int* in_sizes, int n_in,
                              void* d_out, int out_size, void* d_ws, size_t ws_size,
                              hipStream_t stream) {
  // input order: x, l0.fwd{w_ih,w_hh,b}, l0.bwd{...}, l1.fwd{...}, l1.bwd{...}
  const float* x = (const float*)d_in[0];

  char* ws = (char*)d_ws;
  const size_t SZ_WIH0 = (size_t)2 * G4 * 256 * 2;        // 1 MB
  const size_t SZ_WIH1 = (size_t)2 * G4 * 512 * 2;        // 2 MB
  const size_t SZ_WHH  = (size_t)4 * G4 * HID * 2;        // 2 MB
  const size_t SZ_Y0   = (size_t)T_SEQ * BATCH * 512 * 2; // 64 MB
  _Float16* wih0 = (_Float16*)(ws);
  _Float16* wih1 = (_Float16*)(ws + SZ_WIH0);
  _Float16* whh  = (_Float16*)(ws + SZ_WIH0 + SZ_WIH1);
  _Float16* y0   = (_Float16*)(ws + SZ_WIH0 + SZ_WIH1 + SZ_WHH);
  _Float16* xw   = (_Float16*)(ws + SZ_WIH0 + SZ_WIH1 + SZ_WHH + SZ_Y0); // 256 MB

  float* out = (float*)d_out;
  float* hn  = out + (size_t)BATCH * T_SEQ * 512;
  float* cn  = hn + (size_t)4 * BATCH * HID;

  auto cvt = [&](int idx, _Float16* dst, int n) {
    cvt_f32_to_f16<<<(n + 255) / 256, 256, 0, stream>>>((const float*)d_in[idx],
                                                        dst, n);
  };
  cvt(1,  wih0,                 G4 * 256);
  cvt(4,  wih0 + G4 * 256,      G4 * 256);
  cvt(7,  wih1,                 G4 * 512);
  cvt(10, wih1 + G4 * 512,      G4 * 512);
  cvt(2,  whh + 0 * G4 * HID,   G4 * HID);
  cvt(5,  whh + 1 * G4 * HID,   G4 * HID);
  cvt(8,  whh + 2 * G4 * HID,   G4 * HID);
  cvt(11, whh + 3 * G4 * HID,   G4 * HID);

  const uint32_t smem_scan = BATCH * HPAD * 2 + 16 * 2 * 2 * 256 * 4; // 82432
  const uint32_t smem_gemm = BATCH * XROWB;                           // 33280

  gemm_xw_kernel<float, 256><<<dim3(T_SEQ, 2), 256, smem_gemm, stream>>>(
      x, wih0, (const float*)d_in[3], (const float*)d_in[6], xw);
  lstm_scan_kernel<0><<<2, 1024, smem_scan, stream>>>(xw, whh, y0, nullptr, hn, cn);

  gemm_xw_kernel<_Float16, 512><<<dim3(T_SEQ, 2), 256, smem_gemm, stream>>>(
      y0, wih1, (const float*)d_in[9], (const float*)d_in[12], xw);
  lstm_scan_kernel<1><<<2, 1024, smem_scan, stream>>>(xw, whh, nullptr, out, hn, cn);
}
